// ConvDeepSet_20916490732177
// MI455X (gfx1250) — matrix operand order repaired and checked by reading, not compiled, not run on hardware
//
#include <hip/hip_runtime.h>

#define N_C    2048
#define T_PTS  16384
#define KSZ    5
#define TILE_T 128

#if defined(__has_builtin)
#if __has_builtin(__builtin_amdgcn_global_load_async_to_lds_b128)
#define USE_ASYNC_LDS 1
#endif
#endif

typedef __attribute__((ext_vector_type(4)))  int    v4i;
typedef __attribute__((ext_vector_type(8)))  __bf16 v8bf;
typedef __attribute__((ext_vector_type(16))) __bf16 v16bf;
typedef __attribute__((ext_vector_type(8)))  float  v8f;

static __device__ __forceinline__ v16bf bf_cat(v8bf lo, v8bf hi) {
  return __builtin_shufflevector(lo, hi, 0,1,2,3,4,5,6,7,8,9,10,11,12,13,14,15);
}

// ---------------------------------------------------------------------------
// Stage 1: E_z / normalization.  norm = [sum K, (sum y*K)/(sum K)] per target.
// Output written t-major into act0 with stride 32 (cols 2..31 zero) + halo.
// ---------------------------------------------------------------------------
__global__ __launch_bounds__(256) void ez_kernel(
    const float* __restrict__ Xc, const float* __restrict__ yc,
    const float* __restrict__ tgt, const float* __restrict__ log_l,
    __bf16* __restrict__ act0)
{
  __shared__ float sx[256];
  __shared__ float sy[256];
  const float inv_l = __expf(-log_l[0]);
  const int ti = blockIdx.x * 256 + threadIdx.x;
  const float st = tgt[ti] * inv_l;
  float s0 = 0.f, s1 = 0.f;
  for (int base = 0; base < N_C; base += 256) {
    sx[threadIdx.x] = Xc[base + threadIdx.x] * inv_l;
    sy[threadIdx.x] = yc[base + threadIdx.x];
    __syncthreads();
#pragma unroll 8
    for (int j = 0; j < 256; ++j) {
      float e = __expf(-0.5f * fabsf(st - sx[j]));  // x_dim==1: cdist == |dx|
      s0 += e;
      s1 = fmaf(sy[j], e, s1);
    }
    __syncthreads();
  }
  const size_t row = (size_t)(ti + 2) * 32;
  act0[row + 0] = (__bf16)s0;
  act0[row + 1] = (__bf16)(s1 / s0);
#pragma unroll
  for (int c = 2; c < 32; ++c) act0[row + c] = (__bf16)0.f;
}

// ---------------------------------------------------------------------------
// Zero the 2-row halos (t = -2,-1,T,T+1) of all activation buffers.
// ---------------------------------------------------------------------------
__global__ void zero_halos(__bf16* a0, __bf16* aA, __bf16* aB) {
  const int tid = threadIdx.x;
  for (int i = tid; i < 4 * 32; i += 256) {
    int row = i >> 5, c = i & 31;
    int rr = (row < 2) ? row : (row + T_PTS);
    a0[(size_t)rr * 32 + c] = (__bf16)0.f;
  }
  for (int i = tid; i < 4 * 256; i += 256) {
    int row = i >> 8, c = i & 255;
    int rr = (row < 2) ? row : (row + T_PTS);
    aA[(size_t)rr * 256 + c] = (__bf16)0.f;
    aB[(size_t)rr * 256 + c] = (__bf16)0.f;
  }
}

// ---------------------------------------------------------------------------
// Pack fp32 conv weights (C_out, C_in, 5) into bf16 WMMA A-fragments.
// Fragment f = (mtile*5 + k)*nkc + kc, stored [lane][16 bf16] (32B/lane).
// A element (m, kk): ISA 16-bit A 16x32 layout:
//   lane = (kk upper half ? 16 : 0) + m ; elem j -> kk = base + (j<8 ? j : j+8)
// ---------------------------------------------------------------------------
__global__ __launch_bounds__(256) void pack_w(
    const float* __restrict__ W, __bf16* __restrict__ Wp,
    int C_out, int C_in, int nkc)
{
  const int f = blockIdx.x;
  const int kc = f % nkc;
  const int mk = f / nkc;
  const int k = mk % KSZ;
  const int mtile = mk / KSZ;
  const int tid = threadIdx.x;
  const int l = tid >> 3;          // lane 0..31
  const int j0 = (tid & 7) * 2;
#pragma unroll
  for (int jj = 0; jj < 2; ++jj) {
    int j = j0 + jj;
    int m = mtile * 16 + (l & 15);
    int kk = ((l >> 4) << 3) + ((j < 8) ? j : j + 8);
    int c = kc * 32 + kk;
    float v = (m < C_out && c < C_in) ? W[((size_t)m * C_in + c) * KSZ + k] : 0.f;
    Wp[(size_t)f * 512 + l * 16 + j] = (__bf16)v;
  }
}

// ---------------------------------------------------------------------------
// Epilogue: bias + relu, store one 16x16 quadrant-row (8 consecutive c / lane).
// ---------------------------------------------------------------------------
template<bool FINAL>
__device__ __forceinline__ void store_tile(
    v8f acc, int mt, int n_mtiles, int C_out, int relu,
    const float* __restrict__ bias, __bf16* __restrict__ out_bf,
    float* __restrict__ out_f32, int t0, int half)
{
  if (mt >= n_mtiles) return;
  const int cbase = mt * 16 + half * 8;
  if (cbase >= C_out) return;
#pragma unroll
  for (int r = 0; r < 8; ++r) {
    float v = acc[r] + bias[cbase + r];
    if (relu) v = fmaxf(v, 0.f);
    acc[r] = v;
  }
  if (FINAL) {
    float4* dp = (float4*)(out_f32 + (size_t)t0 * C_out + cbase);
    dp[0] = make_float4(acc[0], acc[1], acc[2], acc[3]);
    dp[1] = make_float4(acc[4], acc[5], acc[6], acc[7]);
  } else {
    v8bf ob;
#pragma unroll
    for (int r = 0; r < 8; ++r) ob[r] = (__bf16)acc[r];
    *(v8bf*)(out_bf + (size_t)(t0 + 2) * 256 + cbase) = ob;
  }
}

// ---------------------------------------------------------------------------
// Conv1d (SAME, K=5) as WMMA GEMM, register-blocked 2 mtile x 2 ntile / wave.
// Activations t-major [t+2][c] (bf16, 2-row halo).  CSTR = input c-stride.
// Block: 256 thr = 8 waves (wy 0..1 x wx 0..3); block tile 64 c_out x 128 t.
// Staging uses GLOBAL_LOAD_ASYNC_TO_LDS_B128 (ASYNCcnt) when available.
// B fragment (32x16): lanes 0-15 rows K=0..7,16..23, lanes 16-31 rows K=8..15,24..31;
// with [t][c] layout each lane's 16 K-values are two 16B LDS runs, 32B apart.
// ---------------------------------------------------------------------------
template<int CSTR, bool FINAL>
__global__ __launch_bounds__(256) void conv_wmma(
    const __bf16* __restrict__ in,
    const __bf16* __restrict__ Wp, const float* __restrict__ bias,
    __bf16* __restrict__ out_bf, float* __restrict__ out_f32,
    int C_out, int relu)
{
  constexpr int NKC = CSTR / 32;
  __shared__ __bf16 s_in[(TILE_T + 4) * CSTR];
  const int tid = threadIdx.x;
  const int tbase = blockIdx.x * TILE_T;

  // contiguous, 16B-aligned global -> LDS staging of rows [tbase-2, tbase+129]
  {
    const v4i* src = (const v4i*)(in + (size_t)tbase * CSTR);
    v4i* dst = (v4i*)s_in;
    constexpr int n16 = ((TILE_T + 4) * CSTR * 2) >> 4;
#ifdef USE_ASYNC_LDS
    for (int i = tid; i < n16; i += 256)
      __builtin_amdgcn_global_load_async_to_lds_b128(
          (v4i*)(src + i), (v4i*)(dst + i), 0, 0);
#else
    for (int i = tid; i < n16; i += 256) dst[i] = src[i];
#endif
    // prefetch next t-tile (emits global_prefetch_b8)
    if (blockIdx.x + 1 < gridDim.x) {
      constexpr int span = (TILE_T * CSTR * 2) / 256;
      __builtin_prefetch(
          (const char*)(in + (size_t)(tbase + TILE_T) * CSTR) + tid * span, 0, 1);
    }
#ifdef USE_ASYNC_LDS
#if __has_builtin(__builtin_amdgcn_s_wait_asynccnt)
    __builtin_amdgcn_s_wait_asynccnt(0);
#else
    asm volatile("s_wait_asynccnt 0" ::: "memory");
#endif
#endif
  }
  __syncthreads();

  const int lane = tid & 31;
  const int wave = tid >> 5;
  const int wx = wave & 3;        // t sub-tile (32 t each)
  const int wy = wave >> 2;       // c_out sub-tile (32 c each)
  const int n_mtiles = (C_out + 15) >> 4;
  const int mtA = blockIdx.y * 4 + wy * 2;
  if (mtA >= n_mtiles) return;
  const int mtB = mtA + 1;
  const int mtBc = (mtB < n_mtiles) ? mtB : mtA;   // clamp for safe A reads

  v8f acc00 = {0,0,0,0,0,0,0,0}, acc01 = {0,0,0,0,0,0,0,0};
  v8f acc10 = {0,0,0,0,0,0,0,0}, acc11 = {0,0,0,0,0,0,0,0};
  const int n    = lane & 15;
  const int half = lane >> 4;
  const int bcol = half << 3;          // B K-split: rows 8h..8h+7 and 16+8h..16+8h+7
  const int r0   = wx * 32 + n;        // LDS row of this lane's t (ntile 0)

#pragma unroll
  for (int k = 0; k < KSZ; ++k) {
#pragma unroll
    for (int kc = 0; kc < NKC; ++kc) {
      const size_t fa = ((size_t)(mtA  * KSZ + k) * NKC + kc) * 512 + lane * 16;
      const size_t fb = ((size_t)(mtBc * KSZ + k) * NKC + kc) * 512 + lane * 16;
      v16bf a0 = bf_cat(*(const v8bf*)(Wp + fa), *(const v8bf*)(Wp + fa + 8));
      v16bf a1 = bf_cat(*(const v8bf*)(Wp + fb), *(const v8bf*)(Wp + fb + 8));
      const __bf16* bp0 = &s_in[(size_t)(r0 + k) * CSTR + kc * 32 + bcol];
      const __bf16* bp1 = bp0 + 16 * CSTR;
      v16bf b0 = bf_cat(*(const v8bf*)bp0, *(const v8bf*)(bp0 + 16));
      v16bf b1 = bf_cat(*(const v8bf*)bp1, *(const v8bf*)(bp1 + 16));
      acc00 = __builtin_amdgcn_wmma_f32_16x16x32_bf16(false, a0, false, b0, (short)0, acc00, false, false);
      acc01 = __builtin_amdgcn_wmma_f32_16x16x32_bf16(false, a0, false, b1, (short)0, acc01, false, false);
      acc10 = __builtin_amdgcn_wmma_f32_16x16x32_bf16(false, a1, false, b0, (short)0, acc10, false, false);
      acc11 = __builtin_amdgcn_wmma_f32_16x16x32_bf16(false, a1, false, b1, (short)0, acc11, false, false);
    }
  }

  const int t0 = tbase + wx * 32 + n;  // ntile 0; ntile 1 at +16
  store_tile<FINAL>(acc00, mtA, n_mtiles, C_out, relu, bias, out_bf, out_f32, t0,      half);
  store_tile<FINAL>(acc01, mtA, n_mtiles, C_out, relu, bias, out_bf, out_f32, t0 + 16, half);
  store_tile<FINAL>(acc10, mtB, n_mtiles, C_out, relu, bias, out_bf, out_f32, t0,      half);
  store_tile<FINAL>(acc11, mtB, n_mtiles, C_out, relu, bias, out_bf, out_f32, t0 + 16, half);
}

// ---------------------------------------------------------------------------
extern "C" void kernel_launch(void* const* d_in, const int* in_sizes, int n_in,
                              void* d_out, int out_size, void* d_ws, size_t ws_size,
                              hipStream_t stream) {
  (void)in_sizes; (void)n_in; (void)out_size; (void)ws_size;
  const float* Xc   = (const float*)d_in[0];
  const float* yc   = (const float*)d_in[1];
  const float* tg   = (const float*)d_in[2];
  const float* logl = (const float*)d_in[3];
  const float* W[5]; const float* B[5];
  for (int i = 0; i < 5; ++i) { W[i] = (const float*)d_in[4 + i]; B[i] = (const float*)d_in[9 + i]; }

  char* ws = (char*)d_ws;
  size_t off = 0;
  auto alloc = [&](size_t bytes) -> char* {
    char* p = ws + off;
    off = (off + bytes + 255) & ~(size_t)255;
    return p;
  };
  __bf16* act0 = (__bf16*)alloc((size_t)(T_PTS + 4) * 32 * 2);
  __bf16* actA = (__bf16*)alloc((size_t)(T_PTS + 4) * 256 * 2);
  __bf16* actB = (__bf16*)alloc((size_t)(T_PTS + 4) * 256 * 2);

  const int Cin[5]  = {2, 256, 256, 256, 256};
  const int Cout[5] = {256, 256, 256, 256, 8};
  __bf16* Wp[5];
  int nkcL[5], nmL[5];
  for (int i = 0; i < 5; ++i) {
    nkcL[i] = (Cin[i] + 31) / 32;
    nmL[i]  = (Cout[i] + 15) / 16;
    Wp[i]   = (__bf16*)alloc((size_t)nmL[i] * KSZ * nkcL[i] * 512 * 2);
  }

  zero_halos<<<1, 256, 0, stream>>>(act0, actA, actB);
  for (int i = 0; i < 5; ++i)
    pack_w<<<nmL[i] * KSZ * nkcL[i], 256, 0, stream>>>(W[i], Wp[i], Cout[i], Cin[i], nkcL[i]);
  ez_kernel<<<T_PTS / 256, 256, 0, stream>>>(Xc, yc, tg, logl, act0);

  const dim3 blk(256);
  const int gx = T_PTS / TILE_T;     // 128
  // layer 0: 2 -> 256 (stride-32 input, one zero-padded K chunk)
  conv_wmma<32,  false><<<dim3(gx, 4), blk, 0, stream>>>(act0, Wp[0], B[0], actA, nullptr, 256, 1);
  // layers 1-3: 256 -> 256
  conv_wmma<256, false><<<dim3(gx, 4), blk, 0, stream>>>(actA, Wp[1], B[1], actB, nullptr, 256, 1);
  conv_wmma<256, false><<<dim3(gx, 4), blk, 0, stream>>>(actB, Wp[2], B[2], actA, nullptr, 256, 1);
  conv_wmma<256, false><<<dim3(gx, 4), blk, 0, stream>>>(actA, Wp[3], B[3], actB, nullptr, 256, 1);
  // layer 4: 256 -> 8, fp32 straight into d_out (T,8) t-major, no relu
  conv_wmma<256, true ><<<dim3(gx, 1), blk, 0, stream>>>(actB, Wp[4], B[4], nullptr, (float*)d_out, 8, 0);
}
